// MultiHeadSelfAttention_63247688401405
// MI455X (gfx1250) — matrix operand order repaired and checked by reading, not compiled, not run on hardware
//
#include <hip/hip_runtime.h>
#include <hip/hip_bf16.h>
#include <stdint.h>

typedef __attribute__((ext_vector_type(16))) __bf16 v16bf;
typedef __attribute__((ext_vector_type(8)))  float  v8f;
typedef __attribute__((ext_vector_type(4)))  __bf16 v4bf;

constexpr int B_ = 2;
constexpr int T_ = 2048;
constexpr int CM = 2048;   // d_model
constexpr int H_ = 16;
constexpr int DH = 128;
constexpr int M_ = B_ * T_;  // 4096 tokens

union F8  { v8f   v; float  e[8]; };
union F16 { v16bf v; __bf16 e[16]; uint4 u[2]; };

__device__ __forceinline__ v8f wmma_bf16(v16bf a, v16bf b, v8f c) {
  // D(16x16 f32) = A(16x32 bf16) x B(32x16 bf16) + C
  return __builtin_amdgcn_wmma_f32_16x16x32_bf16(false, a, false, b, (short)0, c,
                                                 false, false);
}

// A fragment 16(M) x 32(K) from row-major [M,K] storage at (row0,k0), leading dim ld.
// ISA layout: lanes 0-15 hold M=lane, K{0..7,16..23}; lanes 16-31 hold K{8..15,24..31}.
__device__ __forceinline__ v16bf load_afrag(const __bf16* base, int ld) {
  int lane = threadIdx.x & 31;
  int m = lane & 15, half = lane >> 4;
  const __bf16* p = base + (size_t)m * ld + half * 8;
  F16 f;
  f.u[0] = *(const uint4*)p;          // K  (+0..7)
  f.u[1] = *(const uint4*)(p + 16);   // K (+16..23)
  return f.v;
}

// B fragment 32(K) x 16(N) from N-major ("transposed") storage [N,K] at (n0,k0).
// ISA layout: lanes 0-15 hold N=lane, K{0..7,16..23}; lanes 16-31 hold K{8..15,24..31}.
__device__ __forceinline__ v16bf load_bfrag(const __bf16* base, int ld) {
  int lane = threadIdx.x & 31;
  int n = lane & 15, half = lane >> 4;
  const __bf16* p = base + (size_t)n * ld + half * 8;
  F16 f;
  f.u[0] = *(const uint4*)p;
  f.u[1] = *(const uint4*)(p + 16);
  return f.v;
}

// Double-buffered bf16 GEMM inner loop shared by both projection kernels.
// acc[2][4] += A(32xK) . B(Kx64) with ping-pong fragment buffers (no reg copies).
#define GEMM_MAIN_LOOP(Arow, Brow, K)                                          \
  v16bf af0_0, af1_0, bf_0[4];                                                 \
  v16bf af0_1, af1_1, bf_1[4];                                                 \
  GLOAD(0, 0);                                                                 \
  GLOAD(1, 32);                                                                \
  for (int k0 = 64; k0 < (K); k0 += 64) {                                      \
    int kp = (k0 + 64) & ((K) - 1);                                            \
    __builtin_prefetch((Arow) + (size_t)ln15 * (K) + kp, 0, 1);                \
    __builtin_prefetch((Brow) + (size_t)(ln15 * 4) * (K) + kp, 0, 1);          \
    GWMMA(0);                                                                  \
    GLOAD(0, k0);                                                              \
    GWMMA(1);                                                                  \
    GLOAD(1, k0 + 32);                                                         \
  }                                                                            \
  GWMMA(0);                                                                    \
  GWMMA(1);

#define GLOAD(sfx, kk)                                                         \
  do {                                                                         \
    af0_##sfx = load_afrag(Arow + (kk), K);                                    \
    af1_##sfx = load_afrag(Arow + (size_t)16 * K + (kk), K);                   \
    _Pragma("unroll") for (int j = 0; j < 4; ++j) bf_##sfx[j] =                \
        load_bfrag(Brow + (size_t)(j * 16) * K + (kk), K);                     \
  } while (0)

#define GWMMA(sfx)                                                             \
  do {                                                                         \
    _Pragma("unroll") for (int j = 0; j < 4; ++j) {                            \
      acc[0][j] = wmma_bf16(af0_##sfx, bf_##sfx[j], acc[0][j]);                \
      acc[1][j] = wmma_bf16(af1_##sfx, bf_##sfx[j], acc[1][j]);                \
    }                                                                          \
  } while (0)

// ---------------- precision conversion kernels ----------------

__global__ void f2bf_kernel(const float* __restrict__ in, __bf16* __restrict__ out,
                            int n4) {
  int i = blockIdx.x * blockDim.x + threadIdx.x;
  if (i < n4) {
    float4 v = reinterpret_cast<const float4*>(in)[i];
    v4bf o = {(__bf16)v.x, (__bf16)v.y, (__bf16)v.z, (__bf16)v.w};
    reinterpret_cast<v4bf*>(out)[i] = o;
  }
}

// in: fp32 [R,C] -> out: bf16 [C,R] (N-major weights for contiguous B-frag loads)
__global__ void tpose_f2bf_kernel(const float* __restrict__ in,
                                  __bf16* __restrict__ out, int R, int C) {
  __shared__ float tile[32][33];
  int tx = threadIdx.x & 31, ty = threadIdx.x >> 5;
  int r0 = blockIdx.y * 32, c0 = blockIdx.x * 32;
#pragma unroll
  for (int i = 0; i < 4; ++i) {
    int r = r0 + ty + i * 8;
    tile[ty + i * 8][tx] = in[(size_t)r * C + c0 + tx];
  }
  __syncthreads();
#pragma unroll
  for (int i = 0; i < 4; ++i) {
    int c = c0 + ty + i * 8;
    out[(size_t)c * R + r0 + tx] = (__bf16)tile[tx][ty + i * 8];
  }
}

// ---------------- QKV projection GEMM + bias + RoPE + layout split ----------------
// C[m,n] = X[m,:] . Wqkv[:,n] ; X bf16 [M,K] row-major, WT bf16 [N,K].
// Block = 128x128 (8 waves, each 32x64). Ping-pong double-buffered k-loop.

__global__ __launch_bounds__(256) void qkv_gemm_kernel(
    const __bf16* __restrict__ Xbf, const __bf16* __restrict__ WT,
    const float* __restrict__ bqkv, const float* __restrict__ fcos,
    const float* __restrict__ fsin, __bf16* __restrict__ q_bf,
    __bf16* __restrict__ k_bf, __bf16* __restrict__ vT_bf,
    float* __restrict__ k_out, float* __restrict__ v_out) {
  const int K = CM;
  int lane = threadIdx.x & 31, wave = threadIdx.x >> 5;
  int ln15 = lane & 15, half = lane >> 4;
  int m_base = blockIdx.y * 128 + (wave >> 1) * 32;
  int n_base = blockIdx.x * 128 + (wave & 1) * 64;

  v8f zero8 = {0, 0, 0, 0, 0, 0, 0, 0};
  v8f acc[2][4];
#pragma unroll
  for (int a = 0; a < 2; ++a)
#pragma unroll
    for (int b = 0; b < 4; ++b) acc[a][b] = zero8;

  const __bf16* Arow = Xbf + (size_t)m_base * K;
  const __bf16* Brow = WT + (size_t)n_base * K;

  GEMM_MAIN_LOOP(Arow, Brow, K)

  // Epilogue: C-tile layout: lane ln15 = column n, VGPR r = row (r + half*8).
#pragma unroll
  for (int mi = 0; mi < 2; ++mi) {
#pragma unroll
    for (int ni = 0; ni < 4; ++ni) {
      F8 c;
      c.v = acc[mi][ni];
      int tn = n_base + ni * 16;
      int nn = tn + ln15;
      int region = nn >> 11;        // 0=q, 1=k, 2=v (each 2048 wide)
      int nl = nn & 2047;
      int h = nl >> 7, d = nl & 127;
      float bias = bqkv[nn];
      int tm = m_base + mi * 16;
#pragma unroll
      for (int r = 0; r < 8; ++r) {
        int mrow = tm + half * 8 + r;
        int b = mrow >> 11, t = mrow & 2047;   // T = 2048
        float v = c.e[r] + bias;
        size_t idx = ((size_t)(b * H_ + h) * T_ + t) * DH + d;
        if (region <= 1) {
          // RoPE: even/odd dim pair lives in adjacent lanes.
          float part = __shfl_xor(v, 1);
          float cs = fcos[t * 64 + (d >> 1)];
          float sn = fsin[t * 64 + (d >> 1)];
          float ro = ((d & 1) == 0) ? (v * cs - part * sn) : (part * sn + v * cs);
          if (region == 0) {
            q_bf[idx] = (__bf16)ro;
          } else {
            k_bf[idx] = (__bf16)ro;
            k_out[idx] = ro;                 // fp32 K output
          }
        } else {
          v_out[idx] = v;                    // fp32 V output
          vT_bf[((size_t)(b * H_ + h) * DH + d) * T_ + t] = (__bf16)v;
        }
      }
    }
  }
}

// ---------------- causal flash attention (bf16 WMMA, f32 online softmax) --------
// grid.x = B*H, grid.y = T/128; each wave owns 16 queries.

__global__ __launch_bounds__(256) void attn_kernel(
    const __bf16* __restrict__ q_bf, const __bf16* __restrict__ k_bf,
    const __bf16* __restrict__ vT_bf, __bf16* __restrict__ attn_bf) {
  int bh = blockIdx.x;
  int b = bh >> 4, h = bh & 15;
  int lane = threadIdx.x & 31, wave = threadIdx.x >> 5;
  int ln15 = lane & 15, half = lane >> 4;
  int q0 = blockIdx.y * 128 + wave * 16;

  const __bf16* Qh = q_bf + (size_t)bh * T_ * DH;
  const __bf16* Kh = k_bf + (size_t)bh * T_ * DH;
  const __bf16* Vt = vT_bf + (size_t)bh * DH * T_;

  // Q as B-operand of S^T = K . Q^T : lane = query column, K-dim = head dims.
  v16bf qf[4];
#pragma unroll
  for (int j = 0; j < 4; ++j) qf[j] = load_bfrag(Qh + (size_t)q0 * DH + j * 32, DH);

  v8f zero8 = {0, 0, 0, 0, 0, 0, 0, 0};
  v8f o[8];
#pragma unroll
  for (int t = 0; t < 8; ++t) o[t] = zero8;

  const float ninf = -__builtin_inff();
  const float scale = 0.08838834764831845f;  // 1/sqrt(128)
  float m_run = ninf, l_run = 0.f;
  int qg = q0 + ln15;  // this lane's query index (as S^T column)

  for (int kb0 = 0; kb0 < q0 + 16; kb0 += 32) {
    // issue all K-tile loads as one batch
    v16bf ka0[4], ka1[4];
#pragma unroll
    for (int j = 0; j < 4; ++j)
      ka0[j] = load_afrag(Kh + (size_t)kb0 * DH + j * 32, DH);
#pragma unroll
    for (int j = 0; j < 4; ++j)
      ka1[j] = load_afrag(Kh + (size_t)(kb0 + 16) * DH + j * 32, DH);

    // S^T tiles: keys as M rows, queries as N cols. K-dim = 128 -> 4 WMMAs each.
    v8f s0 = zero8, s1 = zero8;
#pragma unroll
    for (int j = 0; j < 4; ++j) s0 = wmma_bf16(ka0[j], qf[j], s0);
#pragma unroll
    for (int j = 0; j < 4; ++j) s1 = wmma_bf16(ka1[j], qf[j], s1);

    // issue V loads now so the softmax VALU work below hides their latency
    v16bf vf[8];
#pragma unroll
    for (int t = 0; t < 8; ++t)
      vf[t] = load_bfrag(Vt + (size_t)(t * 16) * T_ + kb0, T_);  // V^T: dim-major

    bool needMask = (kb0 + 31 > q0);
    F8 a0, a1;
    a0.v = s0;
    a1.v = s1;
    float mloc = ninf;
#pragma unroll
    for (int r = 0; r < 8; ++r) {
      float x0 = a0.e[r] * scale;
      float x1 = a1.e[r] * scale;
      if (needMask) {
        if (kb0 + half * 8 + r > qg) x0 = ninf;
        if (kb0 + 16 + half * 8 + r > qg) x1 = ninf;
      }
      a0.e[r] = x0;
      a1.e[r] = x1;
      mloc = fmaxf(mloc, fmaxf(x0, x1));
    }
    mloc = fmaxf(mloc, __shfl_xor(mloc, 16));   // combine the two key halves
    float m_new = fmaxf(m_run, mloc);
    float alpha = __expf(m_run - m_new);
    float lsum = 0.f;
    F16 pa;  // P packed directly into A-frag layout: tile0->V0..3, tile1->V4..7
#pragma unroll
    for (int r = 0; r < 8; ++r) {
      float p0 = __expf(a0.e[r] - m_new);
      float p1 = __expf(a1.e[r] - m_new);
      lsum += p0 + p1;
      pa.e[r] = (__bf16)p0;
      pa.e[8 + r] = (__bf16)p1;
    }
    lsum += __shfl_xor(lsum, 16);
    l_run = l_run * alpha + lsum;
    m_run = m_new;

    // broadcast per-query rescale factor to the O-tile row layout
    float arow[8];
#pragma unroll
    for (int r = 0; r < 8; ++r) arow[r] = __shfl(alpha, half * 8 + r);

#pragma unroll
    for (int t = 0; t < 8; ++t) {
      F8 oo;
      oo.v = o[t];
#pragma unroll
      for (int r = 0; r < 8; ++r) oo.e[r] *= arow[r];
      o[t] = wmma_bf16(pa.v, vf[t], oo.v);
    }
  }

  float linv = 1.f / l_run;
  float lrow[8];
#pragma unroll
  for (int r = 0; r < 8; ++r) lrow[r] = __shfl(linv, half * 8 + r);

  // store merged-head bf16 activation for the output projection
#pragma unroll
  for (int t = 0; t < 8; ++t) {
    F8 oo;
    oo.v = o[t];
#pragma unroll
    for (int r = 0; r < 8; ++r) {
      int token = q0 + half * 8 + r;
      attn_bf[((size_t)(b * T_ + token)) * CM + h * DH + t * 16 + ln15] =
          (__bf16)(oo.e[r] * lrow[r]);
    }
  }
}

// ---------------- output projection GEMM + bias (fp32 out) ----------------

__global__ __launch_bounds__(256) void proj_gemm_kernel(
    const __bf16* __restrict__ Abf, const __bf16* __restrict__ WT,
    const float* __restrict__ bias, float* __restrict__ out) {
  const int K = CM, N = CM;
  int lane = threadIdx.x & 31, wave = threadIdx.x >> 5;
  int ln15 = lane & 15, half = lane >> 4;
  int m_base = blockIdx.y * 128 + (wave >> 1) * 32;
  int n_base = blockIdx.x * 128 + (wave & 1) * 64;

  v8f zero8 = {0, 0, 0, 0, 0, 0, 0, 0};
  v8f acc[2][4];
#pragma unroll
  for (int a = 0; a < 2; ++a)
#pragma unroll
    for (int b = 0; b < 4; ++b) acc[a][b] = zero8;

  const __bf16* Arow = Abf + (size_t)m_base * K;
  const __bf16* Brow = WT + (size_t)n_base * K;

  GEMM_MAIN_LOOP(Arow, Brow, K)

#pragma unroll
  for (int mi = 0; mi < 2; ++mi) {
#pragma unroll
    for (int ni = 0; ni < 4; ++ni) {
      F8 c;
      c.v = acc[mi][ni];
      int nn = n_base + ni * 16 + ln15;
      float bv = bias[nn];
      int tm = m_base + mi * 16;
#pragma unroll
      for (int r = 0; r < 8; ++r) {
        int mrow = tm + half * 8 + r;
        out[(size_t)mrow * N + nn] = c.e[r] + bv;
      }
    }
  }
}

// ---------------- launch ----------------

extern "C" void kernel_launch(void* const* d_in, const int* in_sizes, int n_in,
                              void* d_out, int out_size, void* d_ws, size_t ws_size,
                              hipStream_t stream) {
  (void)in_sizes; (void)n_in; (void)out_size; (void)ws_size;
  const float* x     = (const float*)d_in[0];
  // d_in[1] = boolean causal mask: computed analytically, not read
  const float* fcos  = (const float*)d_in[2];
  const float* fsin  = (const float*)d_in[3];
  const float* Wqkv  = (const float*)d_in[4];
  const float* bqkv  = (const float*)d_in[5];
  const float* Wproj = (const float*)d_in[6];
  const float* bproj = (const float*)d_in[7];

  float* out_p = (float*)d_out;                       // [B,T,C]
  float* k_out = out_p + (size_t)M_ * CM;             // [B,H,T,D]
  float* v_out = k_out + (size_t)B_ * H_ * T_ * DH;   // [B,H,T,D]

  char* ws = (char*)d_ws;
  size_t off = 0;
  auto alloc = [&](size_t bytes) {
    char* p = ws + off;
    off += (bytes + 255) & ~(size_t)255;
    return p;
  };
  __bf16* x_bf    = (__bf16*)alloc((size_t)M_ * CM * 2);
  __bf16* WqkvT   = (__bf16*)alloc((size_t)3 * CM * CM * 2);
  __bf16* WprojT  = (__bf16*)alloc((size_t)CM * CM * 2);
  __bf16* q_bf    = (__bf16*)alloc((size_t)M_ * CM * 2);
  __bf16* k_bf    = (__bf16*)alloc((size_t)M_ * CM * 2);
  __bf16* vT_bf   = (__bf16*)alloc((size_t)M_ * CM * 2);
  __bf16* attn_bf = (__bf16*)alloc((size_t)M_ * CM * 2);

  {
    int n4 = M_ * CM / 4;
    f2bf_kernel<<<(n4 + 255) / 256, 256, 0, stream>>>(x, x_bf, n4);
  }
  tpose_f2bf_kernel<<<dim3(3 * CM / 32, CM / 32), 256, 0, stream>>>(Wqkv, WqkvT,
                                                                    CM, 3 * CM);
  tpose_f2bf_kernel<<<dim3(CM / 32, CM / 32), 256, 0, stream>>>(Wproj, WprojT,
                                                                CM, CM);
  qkv_gemm_kernel<<<dim3(3 * CM / 128, M_ / 128), 256, 0, stream>>>(
      x_bf, WqkvT, bqkv, fcos, fsin, q_bf, k_bf, vT_bf, k_out, v_out);
  attn_kernel<<<dim3(B_ * H_, T_ / 128), 256, 0, stream>>>(q_bf, k_bf, vT_bf,
                                                           attn_bf);
  proj_gemm_kernel<<<dim3(CM / 128, M_ / 128), 256, 0, stream>>>(attn_bf, WprojT,
                                                                 bproj, out_p);
}
